// QPLEXQattenWeight_4329327034501
// MI455X (gfx1250) — compile-verified, matrix operand order checked by reading, not run on hardware
//
#include <hip/hip_runtime.h>
#include <hip/hip_bf16.h>

typedef __bf16 bf16;
typedef __attribute__((ext_vector_type(16))) __bf16 v16bf;
typedef __attribute__((ext_vector_type(8)))  __bf16 v8bf;
typedef __attribute__((ext_vector_type(8)))  float  v8f;

#define B_TOT   16384
#define A_N     32
#define U_N     64
#define S_N     2048
#define E_N     32
#define HYP_N   64
#define H_N     4
#define NCOLS   288                 // 256 (H*HYP) + 32 (v hidden)
#define NT      18                  // N tiles of 16
#define KSTEPS  (S_N / 32)          // 64
#define ROWS    16                  // batch rows per block
#define THREADS 288                 // 9 waves * 32; each wave owns exactly 2 N-tiles
#define NBLOCKS (B_TOT / ROWS)      // 1024
#define BFRAG_ELEMS ((size_t)NT * KSTEPS * 32 * 16)   // 589824 bf16

// ---------------------------------------------------------------------------
// Pack fused weight matrix W[S_N][NCOLS] into bf16 WMMA B-fragments.
// Fragment layout (per ISA 7.12.2, 16-bit operand, wave32):
//   elem j<8  -> K = ks*32 + (lane>>4)*8 + j
//   elem j>=8 -> K = ks*32 + (lane>>4)*8 + 16 + (j-8)
//   N = nt*16 + (lane&15)
// ---------------------------------------------------------------------------
__global__ void qplex_pack_weights(const float* __restrict__ sel_w1,
                                   const float* __restrict__ v_w1,
                                   bf16* __restrict__ Bfrag) {
    int idx = blockIdx.x * blockDim.x + threadIdx.x;
    if (idx >= (int)BFRAG_ELEMS) return;
    int j    = idx & 15;
    int lane = (idx >> 4) & 31;
    int rest = idx >> 9;
    int ks   = rest & (KSTEPS - 1);
    int nt   = rest >> 6;
    int n     = nt * 16 + (lane & 15);
    int kbase = ks * 32 + ((lane >> 4) << 3);
    int k     = (j < 8) ? (kbase + j) : (kbase + 16 + (j - 8));
    float w;
    if (n < H_N * HYP_N) {
        int h = n >> 6, hyp = n & 63;                       // sel_b1-compatible order
        w = sel_w1[(size_t)h * S_N * HYP_N + (size_t)k * HYP_N + hyp];
    } else {
        w = v_w1[(size_t)k * E_N + (n - H_N * HYP_N)];
    }
    Bfrag[idx] = (bf16)w;
}

// ---------------------------------------------------------------------------
// Main fused kernel: 16 batch rows per block, 9 waves, 2 N-tiles per wave.
// GEMM [16 x 2048] x [2048 x 288] with v_wmma_f32_16x16x32_bf16, then
// per-row epilogue (selectors -> m -> logits -> softmax/entropy/reg).
// No divergent control flow in the WMMA loop: EXEC stays all-ones.
// ---------------------------------------------------------------------------
__global__ void __launch_bounds__(THREADS)
qplex_main(const float* __restrict__ states,
           const float* __restrict__ sel_b1,    // [H*HYP] flat == column bias
           const float* __restrict__ sel_w2,    // [H][HYP][E]
           const float* __restrict__ key_w,     // [H][U][E]
           const float* __restrict__ v_b1,      // [E]
           const float* __restrict__ v_w2,      // [E]
           const float* __restrict__ v_b2,      // [1]
           const bf16*  __restrict__ Bfrag,
           float* __restrict__ part,            // [NBLOCKS][8]
           float* __restrict__ out) {
    __shared__ bf16  sA[ROWS][S_N];             // 64 KB bf16 states tile
    __shared__ float sH[ROWS][NCOLS + 4];       // post-ReLU hidden (h1 | vh)
    __shared__ float sSel[ROWS][H_N][E_N];
    __shared__ float sM[ROWS][H_N][U_N];
    __shared__ float sLg[ROWS][H_N][A_N];
    __shared__ float sWt[ROWS][H_N][A_N];
    __shared__ float sRed[64];
    __shared__ float sEnt[64];

    const int tid  = threadIdx.x;
    const int row0 = blockIdx.x * ROWS;

    // --- load states tile, convert fp32 -> bf16 (coalesced) ---
    const float* srow = states + (size_t)row0 * S_N;
    for (int i = tid; i < ROWS * S_N; i += THREADS) {
        int r = i >> 11;            // S_N == 2048
        int c = i & (S_N - 1);
        sA[r][c] = (bf16)srow[(size_t)r * S_N + c];
    }
    __syncthreads();

    // --- WMMA main loop: wave w owns N-tiles (2w, 2w+1); no branches ---
    const int wave = tid >> 5;
    const int lane = tid & 31;
    const int rowA = lane & 15;
    const int koff = (lane >> 4) << 3;
    const int nt0  = wave * 2;
    const int nt1  = wave * 2 + 1;

    // per-ks fragment stride = 32 lanes * 16 elems = 512 bf16
    const bf16* bbase0 = Bfrag + ((size_t)nt0 * KSTEPS) * 512 + (size_t)lane * 16;
    const bf16* bbase1 = Bfrag + ((size_t)nt1 * KSTEPS) * 512 + (size_t)lane * 16;

    v16bf b0 = *(const v16bf*)bbase0;           // prefetch ks = 0
    v16bf b1 = *(const v16bf*)bbase1;
    v8f acc0 = {}, acc1 = {};

    for (int ks = 0; ks < KSTEPS; ++ks) {
        // branchless prefetch of next K-step (wraps to 0 on last iter; harmless)
        int nks = (ks + 1) & (KSTEPS - 1);
        v16bf nb0 = *(const v16bf*)(bbase0 + (size_t)nks * 512);
        v16bf nb1 = *(const v16bf*)(bbase1 + (size_t)nks * 512);

        // A fragment: two 16-byte LDS vector loads
        const bf16* ap = &sA[rowA][ks * 32 + koff];
        v8bf lo = *(const v8bf*)ap;
        v8bf hi = *(const v8bf*)(ap + 16);
        v16bf afrag;
#pragma unroll
        for (int j = 0; j < 8; ++j) {
            afrag[j]     = lo[j];
            afrag[j + 8] = hi[j];
        }

        acc0 = __builtin_amdgcn_wmma_f32_16x16x32_bf16(
            false, afrag, false, b0, (short)0, acc0, false, false);
        acc1 = __builtin_amdgcn_wmma_f32_16x16x32_bf16(
            false, afrag, false, b1, (short)0, acc1, false, false);

        b0 = nb0;
        b1 = nb1;
    }

    // --- bias + ReLU, scatter D fragments to sH (no branches) ---
    {
        const int rbase = (lane >> 4) << 3;     // D row = r + 8*(lane>>4)
        int col0 = nt0 * 16 + (lane & 15);
        int col1 = nt1 * 16 + (lane & 15);
        float bias0 = (col0 < H_N * HYP_N) ? sel_b1[col0] : v_b1[col0 - H_N * HYP_N];
        float bias1 = (col1 < H_N * HYP_N) ? sel_b1[col1] : v_b1[col1 - H_N * HYP_N];
#pragma unroll
        for (int r = 0; r < 8; ++r) {
            float x0 = acc0[r] + bias0;
            float x1 = acc1[r] + bias1;
            sH[rbase + r][col0] = x0 > 0.f ? x0 : 0.f;
            sH[rbase + r][col1] = x1 > 0.f ? x1 : 0.f;
        }
    }
    __syncthreads();

    // --- selectors: sel[row][h][e] = sum_k h1[row][h*64+k] * sel_w2[h][k][e] ---
    for (int o = tid; o < ROWS * H_N * E_N; o += THREADS) {   // 2048 outputs
        int row = o >> 7, rem = o & 127, h = rem >> 5, e = rem & 31;
        const float* w2 = sel_w2 + ((size_t)h * HYP_N) * E_N + e;
        float s = 0.f;
#pragma unroll 8
        for (int k = 0; k < HYP_N; ++k) s += sH[row][h * HYP_N + k] * w2[(size_t)k * E_N];
        sSel[row][h][e] = s;
    }
    __syncthreads();

    // --- m[row][h][u] = sum_e key_w[h][u][e] * sel[row][h][e] ---
    for (int o = tid; o < ROWS * H_N * U_N; o += THREADS) {   // 4096 outputs
        int row = o >> 8, rem = o & 255, h = rem >> 6, u = rem & 63;
        const float* kw = key_w + ((size_t)h * U_N + u) * E_N;
        float s = 0.f;
#pragma unroll
        for (int e = 0; e < E_N; ++e) s += kw[e] * sSel[row][h][e];
        sM[row][h][u] = s;
    }
    __syncthreads();

    // --- logits[row][h][a] = sum_u states[row][a*64+u] * m[row][h][u] ---
    for (int o = tid; o < ROWS * H_N * A_N; o += THREADS) {   // 2048 outputs
        int row = o >> 7, rem = o & 127, h = rem >> 5, a = rem & 31;
        const bf16* us = &sA[row][a * U_N];
        const float* m = &sM[row][h][0];
        float s = 0.f;
#pragma unroll 8
        for (int u = 0; u < U_N; ++u) s += (float)us[u] * m[u];
        sLg[row][h][a] = s;
    }
    __syncthreads();

    // --- softmax / entropy / reg partials: one thread per (row, head) ---
    if (tid < 64) {
        int row = tid >> 2, h = tid & 3;
        const float inv_sqrtE = 0.17677669529663687f; // 1/sqrt(32)
        float myreg = 0.f, mx = -3.4e38f;
        for (int a = 0; a < A_N; ++a) {
            float l = sLg[row][h][a];
            myreg += l * l;
            mx = fmaxf(mx, l * inv_sqrtE);
        }
        float sum = 0.f;
        for (int a = 0; a < A_N; ++a) sum += __expf(sLg[row][h][a] * inv_sqrtE - mx);
        float inv = 1.f / sum;
        float myent = 0.f;
        for (int a = 0; a < A_N; ++a) {
            float w = __expf(sLg[row][h][a] * inv_sqrtE - mx) * inv;
            sWt[row][h][a] = w;
            myent -= w * __logf(w + 1e-8f);
        }
        sRed[tid] = myreg;
        sEnt[tid] = myent;
    }
    __syncthreads();

    if (tid < 4) {                                    // per-head entropy partial
        float e = 0.f;
        for (int row = 0; row < ROWS; ++row) e += sEnt[row * 4 + tid];
        part[(size_t)blockIdx.x * 8 + 1 + tid] = e;
    }
    if (tid == 0) {                                   // reg partial
        float r = 0.f;
        for (int i = 0; i < 64; ++i) r += sRed[i];
        part[(size_t)blockIdx.x * 8] = r;
    }

    // --- head_attend[b][a] = sum_h w[b][h][a] ---
    for (int o = tid; o < ROWS * A_N; o += THREADS) {         // 512 outputs
        int row = o >> 5, a = o & 31;
        float s = sWt[row][0][a] + sWt[row][1][a] + sWt[row][2][a] + sWt[row][3][a];
        out[(size_t)(row0 + row) * A_N + a] = s;
    }

    // --- v[b] = vh . v_w2 + v_b2 ---
    if (tid < ROWS) {
        float s = 0.f;
#pragma unroll
        for (int e = 0; e < E_N; ++e) s += sH[tid][H_N * HYP_N + e] * v_w2[e];
        out[(size_t)B_TOT * A_N + row0 + tid] = s + v_b2[0];
    }
}

// ---------------------------------------------------------------------------
// Deterministic final reduction of per-block partials.
// ---------------------------------------------------------------------------
__global__ void qplex_finalize(const float* __restrict__ part, float* __restrict__ out) {
    int t = threadIdx.x;
    const size_t base = (size_t)B_TOT * A_N + B_TOT;
    if (t == 0) {
        float r = 0.f;
        for (int i = 0; i < NBLOCKS; ++i) r += part[(size_t)i * 8];
        out[base] = 1e-3f * r / (float)((size_t)B_TOT * A_N);
    } else if (t <= 4) {
        int h = t - 1;
        float e = 0.f;
        for (int i = 0; i < NBLOCKS; ++i) e += part[(size_t)i * 8 + 1 + h];
        out[base + 1 + h] = e / (float)B_TOT;
    }
}

extern "C" void kernel_launch(void* const* d_in, const int* in_sizes, int n_in,
                              void* d_out, int out_size, void* d_ws, size_t ws_size,
                              hipStream_t stream) {
    // setup_inputs() order:
    // 0 agent_qs (unused by reference outputs), 1 states, 2 sel_w1, 3 sel_b1,
    // 4 sel_w2, 5 key_w, 6 v_w1, 7 v_b1, 8 v_w2, 9 v_b2
    const float* states = (const float*)d_in[1];
    const float* sel_w1 = (const float*)d_in[2];
    const float* sel_b1 = (const float*)d_in[3];
    const float* sel_w2 = (const float*)d_in[4];
    const float* key_w  = (const float*)d_in[5];
    const float* v_w1   = (const float*)d_in[6];
    const float* v_b1   = (const float*)d_in[7];
    const float* v_w2   = (const float*)d_in[8];
    const float* v_b2   = (const float*)d_in[9];
    float* out = (float*)d_out;

    bf16*  Bfrag = (bf16*)d_ws;
    float* part  = (float*)((char*)d_ws + BFRAG_ELEMS * sizeof(bf16));

    qplex_pack_weights<<<(int)((BFRAG_ELEMS + 255) / 256), 256, 0, stream>>>(
        sel_w1, v_w1, Bfrag);
    qplex_main<<<NBLOCKS, THREADS, 0, stream>>>(
        states, sel_b1, sel_w2, key_w, v_b1, v_w2, v_b2, Bfrag, part, out);
    qplex_finalize<<<1, 32, 0, stream>>>(part, out);
}
